// RNNWrapper_18786186953256
// MI455X (gfx1250) — compile-verified, matrix operand order checked by reading, not run on hardware
//
#include <hip/hip_runtime.h>
#include <math.h>

// ---------------------------------------------------------------------------
// RNN language model head for MI455X (gfx1250, wave32, WMMA).
//   Phase 0: pack W_h fp32 [1024,1536] -> bf16 WMMA B-fragments in d_ws (L2-resident, 3 MB)
//   Phase 1: 4 persistent workgroups (one per 16-row batch block) run the 512-step
//            recurrence with v_wmma_f32_16x16x32_bf16, z=[h,x_t] staged in LDS.
//            Embedding rows for step t+1 are register-prefetched during the WMMA
//            loop so the inter-barrier write phase has no global-load latency.
//   Phase 2: logits = h_final @ W_o.T + b_o with v_wmma_f32_16x16x4_f32 (fp32 exact,
//            HBM-bound streaming of 206 MB W_o), written to d_out.
//   Phase 3: row softmax in-place on d_out.
// ---------------------------------------------------------------------------

typedef __attribute__((ext_vector_type(16))) __bf16 v16bf;
typedef __attribute__((ext_vector_type(8)))  float  v8f;
typedef __attribute__((ext_vector_type(2)))  float  v2f;

#define VOCAB 50257
#define EMB   512
#define HID   1024
#define OUTD  50257
#define BATCH 64
#define TLEN  512

#define ZK       (HID + EMB)     // 1536  (concat [h, x])
#define KCH      (ZK / 32)       // 48 bf16 k-chunks
#define ZPAD     8               // halves of pad per row -> row stride 3088 B (banks rotate)
#define ZSTRIDE  (ZK + ZPAD)     // 1544 halves

#if __has_builtin(__builtin_amdgcn_tanhf)
static __device__ __forceinline__ float fast_tanh(float x) { return __builtin_amdgcn_tanhf(x); }
#else
static __device__ __forceinline__ float fast_tanh(float x) { return tanhf(x); }
#endif

static __device__ __forceinline__ unsigned short f32_to_bf16(float f) {
    unsigned int u = __builtin_bit_cast(unsigned int, f);
    unsigned int r = u + 0x7FFFu + ((u >> 16) & 1u);       // round-to-nearest-even
    if ((u & 0x7F800000u) == 0x7F800000u) r = u;           // keep NaN/Inf payload
    return (unsigned short)(r >> 16);
}

static __device__ __forceinline__ unsigned int pack_bf16x2(float lo, float hi) {
    return (unsigned int)f32_to_bf16(lo) | ((unsigned int)f32_to_bf16(hi) << 16);
}

// --------------------------------------------------------------------------
// Phase 0: pack W_h into per-(kc,nt) fragments; element order matches the
// lane/VGPR layout we consume with v_wmma_f32_16x16x32_bf16 (B = 32x16):
// lane l: N = nt*16 + (l&15), halves j -> K = (j<8 ? j : j+8) + 8*(l>=16).
// Each lane's 16 halves are 32 contiguous bytes -> one v16bf global load.
// --------------------------------------------------------------------------
__global__ __launch_bounds__(256) void pack_Wh_kernel(const float* __restrict__ Wh,
                                                      unsigned short* __restrict__ Wp) {
    const int lane = threadIdx.x & 31;
    const int wave = threadIdx.x >> 5;
    const int frag = blockIdx.x * 8 + wave;    // frag = kc*64 + nt, 0..3071
    const int kc = frag >> 6;
    const int nt = frag & 63;
    const int n   = nt * 16 + (lane & 15);
    const int khi = (lane >> 4) * 8;
    unsigned long long base = ((unsigned long long)frag * 32u + (unsigned)lane) * 16u;
    #pragma unroll
    for (int j = 0; j < 16; ++j) {
        int klocal = (j < 8 ? j : j + 8) + khi;
        int k = kc * 32 + klocal;
        Wp[base + j] = f32_to_bf16(Wh[n * ZK + k]);
    }
}

// --------------------------------------------------------------------------
// Phase 1: persistent recurrence. 512 threads = 16 waves; wave w owns N-tiles
// 4w..4w+3. A-fragment (z rows, 16x32 bf16) read from LDS with 2x ds_load_b128
// per lane following the ISA 16-bit A layout. Two barriers per step.
// --------------------------------------------------------------------------
__global__ __launch_bounds__(512) void rnn_scan_kernel(const int*   __restrict__ tok,
                                                       const float* __restrict__ embed,
                                                       const unsigned short* __restrict__ Wp,
                                                       const float* __restrict__ bh,
                                                       const float* __restrict__ h0,
                                                       float* __restrict__ hfin) {
    __shared__ __align__(16) unsigned short Z[16 * ZSTRIDE];   // ~49.4 KB

    const int tid  = threadIdx.x;
    const int lane = tid & 31;
    const int wave = tid >> 5;
    const int blk  = blockIdx.x;                 // batch rows blk*16 .. blk*16+15
    const int r16  = tid >> 5;                   // row 0..15 for fill work
    const int part = tid & 31;

    // ---- init Z: cols [0,HID) = hidden0 (broadcast), cols [HID,ZK) = embed(t=0)
    {
        #pragma unroll
        for (int j = 0; j < 32; j += 2) {
            int c = part * 32 + j;
            *(unsigned int*)((char*)Z + r16 * (ZSTRIDE * 2) + c * 2) =
                pack_bf16x2(h0[c], h0[c + 1]);
        }
        int token = tok[(blk * 16 + r16) * TLEN + 0];
        const float* er = embed + (long long)token * EMB;
        #pragma unroll
        for (int j = 0; j < 16; j += 2) {
            int c = part * 16 + j;
            *(unsigned int*)((char*)Z + r16 * (ZSTRIDE * 2) + (HID + c) * 2) =
                pack_bf16x2(er[c], er[c + 1]);
        }
    }
    __syncthreads();

    float bias[4];
    #pragma unroll
    for (int i = 0; i < 4; ++i) bias[i] = bh[(wave * 4 + i) * 16 + (lane & 15)];

    // A fragment byte base: row = lane%16; lanes>=16 shifted +16 B (K 8..15 / 24..31)
    const int abyte0 = (lane & 15) * (ZSTRIDE * 2) + (lane >> 4) * 16;

    union AFrag { struct { uint4 lo; uint4 hi; } q; v16bf v; };

    for (int t = 0; t < TLEN; ++t) {
        // ---- issue the embedding prefetch for step t+1 FIRST so the global
        //      loads overlap the entire WMMA loop (waited on only after the
        //      barrier, when we convert+store to LDS).
        float4 epf[4];
        if (t < TLEN - 1) {
            int token = tok[(blk * 16 + r16) * TLEN + (t + 1)];
            const float4* er4 = (const float4*)(embed + (long long)token * EMB + part * 16);
            #pragma unroll
            for (int q = 0; q < 4; ++q) epf[q] = er4[q];
        }

        v8f acc[4];
        #pragma unroll
        for (int i = 0; i < 4; ++i) acc[i] = (v8f){0.f,0.f,0.f,0.f,0.f,0.f,0.f,0.f};

        for (int kc = 0; kc < KCH; ++kc) {
            AFrag A;
            const char* ap = (const char*)Z + abyte0 + kc * 64;
            A.q.lo = *(const uint4*)(ap);            // K 0..7   (or 8..15)
            A.q.hi = *(const uint4*)(ap + 32);       // K 16..23 (or 24..31)
            const v16bf* wp = (const v16bf*)Wp + (((long long)kc * 64 + wave * 4) * 32 + lane);
            #pragma unroll
            for (int i = 0; i < 4; ++i) {
                v16bf b = wp[i * 32];
                acc[i] = __builtin_amdgcn_wmma_f32_16x16x32_bf16(
                    false, A.v, false, b, (short)0, acc[i], false, false);
            }
        }

        // bias + tanh in fp32 (C/D layout: VGPR r -> M = r + 8*(lane>=16), N = lane%16)
        #pragma unroll
        for (int i = 0; i < 4; ++i)
            #pragma unroll
            for (int r = 0; r < 8; ++r)
                acc[i][r] = fast_tanh(acc[i][r] + bias[i]);

        __syncthreads();   // everyone finished reading Z

        const int mbase = (lane >> 4) * 8;
        if (t == TLEN - 1) {
            #pragma unroll
            for (int i = 0; i < 4; ++i) {
                int n = (wave * 4 + i) * 16 + (lane & 15);
                #pragma unroll
                for (int r = 0; r < 8; ++r)
                    hfin[(long long)(blk * 16 + mbase + r) * HID + n] = acc[i][r];
            }
        } else {
            // write h_new (bf16) into Z cols [0,HID)
            #pragma unroll
            for (int i = 0; i < 4; ++i) {
                int n = (wave * 4 + i) * 16 + (lane & 15);
                #pragma unroll
                for (int r = 0; r < 8; ++r)
                    Z[(mbase + r) * ZSTRIDE + n] = f32_to_bf16(acc[i][r]);
            }
            // store prefetched embedding row for t+1 into cols [HID,ZK)
            char* zrow = (char*)Z + r16 * (ZSTRIDE * 2) + (HID + part * 16) * 2;
            #pragma unroll
            for (int q = 0; q < 4; ++q) {
                *(unsigned int*)(zrow + q * 8 + 0) = pack_bf16x2(epf[q].x, epf[q].y);
                *(unsigned int*)(zrow + q * 8 + 4) = pack_bf16x2(epf[q].z, epf[q].w);
            }
        }
        __syncthreads();   // Z ready for next step
    }
}

// --------------------------------------------------------------------------
// Phase 2: logits with fp32 WMMA 16x16x4 (exact, HBM-bound on W_o stream).
// Grid: x = N-tile groups (32 tiles/block), y = 4 batch blocks. 8 waves/block,
// each wave does 4 adjacent vocab tiles sharing the A fragment.
// --------------------------------------------------------------------------
__global__ __launch_bounds__(256) void logits_kernel(const float* __restrict__ hfin,
                                                     const float* __restrict__ Wo,
                                                     const float* __restrict__ bo,
                                                     float* __restrict__ out) {
    const int lane = threadIdx.x & 31;
    const int wave = threadIdx.x >> 5;
    const int mblk = blockIdx.y;
    const int nt0  = blockIdx.x * 32 + wave * 4;

    // A frag: lane -> M = lane%16, K pair = kc*4 + 2*(lane>=16)
    const float* hrow = hfin + (long long)(mblk * 16 + (lane & 15)) * HID + (lane >> 4) * 2;

    int ocol[4];
    const float* wrow[4];
    #pragma unroll
    for (int i = 0; i < 4; ++i) {
        int o = (nt0 + i) * 16 + (lane & 15);
        ocol[i] = o;
        int oc = o < OUTD ? o : OUTD - 1;                // clamp: col o only affects output o
        wrow[i] = Wo + (long long)oc * HID + (lane >> 4) * 2;
    }

    v8f acc[4];
    #pragma unroll
    for (int i = 0; i < 4; ++i) acc[i] = (v8f){0.f,0.f,0.f,0.f,0.f,0.f,0.f,0.f};

    #pragma unroll 4
    for (int kc = 0; kc < HID / 4; ++kc) {               // 256 k-chunks of 4
        v2f a = *(const v2f*)(hrow + kc * 4);
        #pragma unroll
        for (int i = 0; i < 4; ++i) {
            v2f b = *(const v2f*)(wrow[i] + kc * 4);
            acc[i] = __builtin_amdgcn_wmma_f32_16x16x4_f32(
                false, a, false, b, (short)0, acc[i], false, false);
        }
    }

    const int mbase = mblk * 16 + 8 * (lane >> 4);
    #pragma unroll
    for (int i = 0; i < 4; ++i) {
        int o = ocol[i];
        if (o < OUTD) {
            float bb = bo[o];
            #pragma unroll
            for (int r = 0; r < 8; ++r)
                out[(long long)(mbase + r) * OUTD + o] = acc[i][r] + bb;
        }
    }
}

// --------------------------------------------------------------------------
// Phase 3: in-place row softmax (max, sum-exp, normalize).
// --------------------------------------------------------------------------
__global__ __launch_bounds__(256) void softmax_kernel(float* __restrict__ out) {
    __shared__ float red[256];
    const int tid = threadIdx.x;
    float* p = out + (long long)blockIdx.x * OUTD;

    float m = -INFINITY;
    for (int i = tid; i < OUTD; i += 256) m = fmaxf(m, p[i]);
    red[tid] = m; __syncthreads();
    for (int s = 128; s > 0; s >>= 1) {
        if (tid < s) red[tid] = fmaxf(red[tid], red[tid + s]);
        __syncthreads();
    }
    m = red[0]; __syncthreads();

    float sum = 0.f;
    for (int i = tid; i < OUTD; i += 256) sum += expf(p[i] - m);
    red[tid] = sum; __syncthreads();
    for (int s = 128; s > 0; s >>= 1) {
        if (tid < s) red[tid] += red[tid + s];
        __syncthreads();
    }
    const float inv = 1.0f / red[0];

    for (int i = tid; i < OUTD; i += 256) p[i] = expf(p[i] - m) * inv;
}

// --------------------------------------------------------------------------
extern "C" void kernel_launch(void* const* d_in, const int* in_sizes, int n_in,
                              void* d_out, int out_size, void* d_ws, size_t ws_size,
                              hipStream_t stream) {
    const int*   tok   = (const int*)d_in[0];     // [64,512]
    const float* embed = (const float*)d_in[1];   // [50257,512]
    const float* Wh    = (const float*)d_in[2];   // [1024,1536]
    const float* bh    = (const float*)d_in[3];   // [1024]
    const float* Wo    = (const float*)d_in[4];   // [50257,1024]
    const float* bo    = (const float*)d_in[5];   // [50257]
    const float* h0    = (const float*)d_in[6];   // [1024]
    float* out = (float*)d_out;                   // [64,50257]

    unsigned short* Wp = (unsigned short*)d_ws;                       // 3,145,728 B packed W_h
    float* hfin = (float*)((char*)d_ws + (size_t)KCH * 64 * 32 * 16 * 2); // 64*1024 fp32

    pack_Wh_kernel<<<384, 256, 0, stream>>>(Wh, Wp);                  // 3072 fragments
    rnn_scan_kernel<<<4, 512, 0, stream>>>(tok, embed, Wp, bh, h0, hfin);

    const int NT = (OUTD + 15) / 16;                                  // 3142
    logits_kernel<<<dim3((NT + 31) / 32, 4), 256, 0, stream>>>(hfin, Wo, bo, out);
    softmax_kernel<<<BATCH, 256, 0, stream>>>(out);
}
